// MyModel_87522843560308
// MI455X (gfx1250) — compile-verified, matrix operand order checked by reading, not run on hardware
//
#include <hip/hip_runtime.h>
#include <hip/hip_bf16.h>

typedef _Float16 v16h __attribute__((ext_vector_type(16)));
typedef float    v8f  __attribute__((ext_vector_type(8)));
typedef float    v4f  __attribute__((ext_vector_type(4)));
typedef float    v2f  __attribute__((ext_vector_type(2)));

// ---------------------------------------------------------------------------
// WMMA lane-layout helpers (CDNA5, wave32, 16x16 tiles; CDNA5 ISA)
// 16-bit A 16x32: lane<16: h[0..7]=K0..7, h[8..15]=K16..23; lane>=16: +8
//   => per k-tile each lane reads 2 contiguous 8-float runs at
//      off0 = kt*32 + (half?8:0) and off0+16.
// C/D f32 16x16: reg i -> M = i + (lane>=16 ? 8 : 0), N = lane&15
// ---------------------------------------------------------------------------
__device__ __forceinline__ int kmap16(int i, int half) {
    return (i < 8 ? i : i + 8) + (half ? 8 : 0);
}

// branch-free A-fragment build from two contiguous 8-float runs
__device__ __forceinline__ v16h cvt16(const float* p0, const float* p1) {
    v4f a0 = *(const v4f*)p0, a1 = *(const v4f*)(p0 + 4);
    v4f b0 = *(const v4f*)p1, b1 = *(const v4f*)(p1 + 4);
    v16h r;
#pragma unroll
    for (int i = 0; i < 4; ++i) {
        r[i]      = (_Float16)a0[i];
        r[4 + i]  = (_Float16)a1[i];
        r[8 + i]  = (_Float16)b0[i];
        r[12 + i] = (_Float16)b1[i];
    }
    return r;
}

// sizes
#define T_POOL 75
#define Bn 64
#define Fc 320
#define Hh 320
#define N3H 960
#define FLATK 48000
#define D1n 2000
#define D2n 301

// packed fragment sizes
#define CONV_KT 4     // K=104 padded to 128 (B frags zero for K>=104)
#define CONV_NT 20    // N=320
#define GRU_KT 10     // K=320
#define GRU_NT 60     // N=960

// ===========================================================================
// Pack conv weights (26,4,320) -> f16 B-fragments (K padded 104->128, zeros)
// frag base = ((nt*KT + kt)*32 + lane)*16 halves
// ===========================================================================
__global__ void pack_conv_w(const float* __restrict__ w, _Float16* __restrict__ out) {
    int idx = blockIdx.x * blockDim.x + threadIdx.x;
    if (idx >= CONV_KT * CONV_NT * 32) return;
    int lane = idx & 31;
    int kt = (idx >> 5) % CONV_KT;
    int nt = (idx >> 5) / CONV_KT;
    int half = (lane >> 4) & 1;
    int n = nt * 16 + (lane & 15);
    _Float16* dst = out + ((size_t)(nt * CONV_KT + kt) * 32 + lane) * 16;
#pragma unroll
    for (int i = 0; i < 16; ++i) {
        int kk = kmap16(i, half) + kt * 32;
        dst[i] = (_Float16)((kk < 104) ? w[kk * Fc + n] : 0.0f);
    }
}

// ===========================================================================
// Pack GRU weights: sel 0=fw_k 1=bw_k 2=fw_rk 3=bw_rk, each (320,960)
// ===========================================================================
__global__ void pack_gru_w(const float* __restrict__ fwk,  const float* __restrict__ bwk,
                           const float* __restrict__ fwrk, const float* __restrict__ bwrk,
                           _Float16* __restrict__ kpack, _Float16* __restrict__ rkpack) {
    int idx = blockIdx.x * blockDim.x + threadIdx.x;
    const int per = GRU_KT * GRU_NT * 32;               // 19200
    if (idx >= 4 * per) return;
    int sel = idx / per, r = idx % per;
    int lane = r & 31;
    int kt = (r >> 5) % GRU_KT;
    int nt = (r >> 5) / GRU_KT;
    int half = (lane >> 4) & 1;
    int n = nt * 16 + (lane & 15);
    const float* src = (sel == 0) ? fwk : (sel == 1) ? bwk : (sel == 2) ? fwrk : bwrk;
    _Float16* base = ((sel < 2) ? kpack : rkpack) + (size_t)(sel & 1) * GRU_KT * GRU_NT * 512;
    _Float16* dst = base + ((size_t)(nt * GRU_KT + kt) * 32 + lane) * 16;
#pragma unroll
    for (int i = 0; i < 16; ++i) {
        int K = kmap16(i, half) + kt * 32;
        dst[i] = (_Float16)src[K * N3H + n];
    }
}

// ===========================================================================
// Conv1D(valid)+bias+ReLU+MaxPool13 fused as im2col GEMM + in-register pool.
// Block = (tp, b), 160 threads = 5 waves; wave w covers f = w*64..w*64+63.
// A rows = 13 conv positions (+3 pad rows address-clamped; their C rows are
// simply discarded by the pooling mask, so no zeroing/branching needed).
// ===========================================================================
__global__ __launch_bounds__(160)
void conv_pool_kernel(const float* __restrict__ in, const float* __restrict__ cb,
                      const _Float16* __restrict__ wpack, float* __restrict__ P) {
    int tp = blockIdx.x, b = blockIdx.y;
    int tid = threadIdx.x;
    int wave = tid >> 5, lane = tid & 31;
    int row = lane & 15, half = (lane >> 4) & 1;
    int tc = tp * 13 + row;
    if (tc > 974) tc = 974;                               // clamp pad rows in-bounds
    const float* patch = in + (size_t)(b * 1000 + tc) * 4; // contiguous 104-float patch
    v8f acc[4] = {};
#pragma unroll
    for (int kt = 0; kt < CONV_KT; ++kt) {
        v16h a;
        if (kt < 3) {
            int off0 = kt * 32 + (half ? 8 : 0);          // all K < 96 < 104: valid
            a = cvt16(patch + off0, patch + off0 + 16);
        } else {
            // only K=96..103 are live (B frags zero beyond); half-lanes' slot is dead
            v4f a0 = *(const v4f*)(patch + 96);
            v4f a1 = *(const v4f*)(patch + 100);
            float s = half ? 0.0f : 1.0f;
#pragma unroll
            for (int i = 0; i < 4; ++i) {
                a[i]      = (_Float16)(a0[i] * s);
                a[4 + i]  = (_Float16)(a1[i] * s);
                a[8 + i]  = (_Float16)0.0f;
                a[12 + i] = (_Float16)0.0f;
            }
        }
#pragma unroll
        for (int j = 0; j < 4; ++j) {
            int nt = wave * 4 + j;
            const v16h* bf = (const v16h*)(wpack + ((size_t)(nt * CONV_KT + kt) * 32 + lane) * 16);
            acc[j] = __builtin_amdgcn_wmma_f32_16x16x32_f16(
                false, a, false, *bf, (short)0, acc[j], false, false);
        }
    }
#pragma unroll
    for (int j = 0; j < 4; ++j) {
        int f = (wave * 4 + j) * 16 + row;
        float bias = cb[f];
        float mx = 0.0f;                       // ReLU floor doubles as pool init
#pragma unroll
        for (int i = 0; i < 8; ++i) {
            int M = i + (half ? 8 : 0);
            if (M < 13) mx = fmaxf(mx, acc[j][i] + bias);
        }
        mx = fmaxf(mx, __shfl_xor(mx, 16, 32));   // combine the two lane halves
        if (!half) P[((size_t)tp * Bn + b) * Fc + f] = mx;   // P is (T,B,F)
    }
}

// ===========================================================================
// GRU input projection: XP[d] = P(4800x320) @ k_d(320x960) + bi_d
// one wave per block: 16(M) x 64(N) tile, K=320
// ===========================================================================
__global__ __launch_bounds__(32)
void xp_gemm(const float* __restrict__ P, const _Float16* __restrict__ kpack,
             const float* __restrict__ fwb, const float* __restrict__ bwb,
             float* __restrict__ XP) {
    int ng = blockIdx.x, mt = blockIdx.y, d = blockIdx.z;
    int lane = threadIdx.x & 31, row = lane & 15, half = (lane >> 4) & 1;
    const float* bi = d ? bwb : fwb;                  // row 0 of (2,3H) bias
    const _Float16* kp = kpack + (size_t)d * GRU_KT * GRU_NT * 512;
    const float* arow = P + (size_t)(mt * 16 + row) * Fc;
    v8f acc[4] = {};
#pragma unroll 2
    for (int kt = 0; kt < GRU_KT; ++kt) {
        int off0 = kt * 32 + (half ? 8 : 0);
        v16h a = cvt16(arow + off0, arow + off0 + 16);
#pragma unroll
        for (int j = 0; j < 4; ++j) {
            int nt = ng * 4 + j;
            const v16h* bf = (const v16h*)(kp + ((size_t)(nt * GRU_KT + kt) * 32 + lane) * 16);
            acc[j] = __builtin_amdgcn_wmma_f32_16x16x32_f16(
                false, a, false, *bf, (short)0, acc[j], false, false);
        }
    }
#pragma unroll
    for (int j = 0; j < 4; ++j) {
        int n = (ng * 4 + j) * 16 + row;
        float bb = bi[n];
#pragma unroll
        for (int i = 0; i < 8; ++i) {
            int m = mt * 16 + i + (half ? 8 : 0);
            XP[((size_t)d * 4800 + m) * N3H + n] = acc[j][i] + bb;
        }
    }
}

// ===========================================================================
// GRU recurrence. 16 blocks: d = bx>>3, 8 batch rows each. 480 thr = 15 waves,
// each wave owns 4 N-tiles of the 8x960 h@rk GEMM per step (f16 WMMA, h in LDS).
// A rows 8..15 are address-clamped duplicates; their C rows are discarded.
// ===========================================================================
__global__ __launch_bounds__(480)
void gru_kernel(const float* __restrict__ XP, const _Float16* __restrict__ rkpack,
                const float* __restrict__ fwb, const float* __restrict__ bwb,
                float* __restrict__ Y) {
    __shared__ __align__(16) float hbuf[8 * Hh];        // 10 KB
    __shared__ __align__(16) float mibuf[8 * N3H];      // 30 KB
    int d = blockIdx.x >> 3;
    int m0 = (blockIdx.x & 7) * 8;
    int tid = threadIdx.x;
    int wave = tid >> 5, lane = tid & 31, row = lane & 15, half = (lane >> 4) & 1;
    int hrow = row & 7;                                  // clamp: rows 8..15 -> dup
    const float* br = (d ? bwb : fwb) + N3H;             // row 1 of (2,3H)
    const _Float16* rk = rkpack + (size_t)d * GRU_KT * GRU_NT * 512;
    const float* xpd = XP + (size_t)d * 4800 * N3H;
    for (int i = tid; i < 8 * Hh; i += 480) hbuf[i] = 0.0f;
    __syncthreads();

    for (int s = 0; s < 75; ++s) {
        int t = d ? (74 - s) : s;
        // mi = h @ rk
#pragma unroll 1
        for (int j = 0; j < 4; ++j) {
            int nt = wave * 4 + j;
            v8f acc = {};
#pragma unroll 2
            for (int kt = 0; kt < GRU_KT; ++kt) {
                int off0 = kt * 32 + (half ? 8 : 0);
                v16h a = cvt16(&hbuf[hrow * Hh + off0], &hbuf[hrow * Hh + off0 + 16]);
                const v16h* bf = (const v16h*)(rk + ((size_t)(nt * GRU_KT + kt) * 32 + lane) * 16);
                acc = __builtin_amdgcn_wmma_f32_16x16x32_f16(
                    false, a, false, *bf, (short)0, acc, false, false);
            }
            if (!half) {
#pragma unroll
                for (int i = 0; i < 8; ++i) mibuf[i * N3H + nt * 16 + row] = acc[i];
            }
        }
        __syncthreads();
        // gates + state update + write output (concat layout)
        for (int idx = tid; idx < 8 * Hh; idx += 480) {
            int m = idx / Hh, jc = idx % Hh;
            int b = m0 + m;
            const float* xp = xpd + ((size_t)t * Bn + b) * N3H;
            float xz = xp[jc], xr = xp[jc + Hh], xh = xp[jc + 2 * Hh];
            float hz = mibuf[m * N3H + jc]           + br[jc];
            float hr = mibuf[m * N3H + jc + Hh]      + br[jc + Hh];
            float hh = mibuf[m * N3H + jc + 2 * Hh]  + br[jc + 2 * Hh];
            float z = 1.0f / (1.0f + __expf(-(xz + hz)));
            float r = 1.0f / (1.0f + __expf(-(xr + hr)));
            float cand = tanhf(xh + r * hh);
            float hn = z * hbuf[idx] + (1.0f - z) * cand;
            hbuf[idx] = hn;
            Y[(size_t)b * FLATK + t * (2 * Hh) + d * Hh + jc] = hn;
        }
        __syncthreads();
    }
}

// ===========================================================================
// Dense1: (64 x 48000) @ w1(48000 x 2000), fp32 WMMA 16x16x4 (w1 streamed fp32,
// memory-roofline term: 384 MB / 23.3 TB/s ~ 16.5 us). K split 8-ways.
// ===========================================================================
__global__ __launch_bounds__(32)
void dense1_kernel(const float* __restrict__ Y, const float* __restrict__ w1,
                   float* __restrict__ part) {
    int nt = blockIdx.x, mt = blockIdx.y, kc = blockIdx.z;
    int lane = threadIdx.x & 31, row = lane & 15, half = (lane >> 4) & 1;
    int n = nt * 16 + row;
    const float* ya = Y + (size_t)(mt * 16 + row) * FLATK + (half ? 2 : 0);
    const float* wb = w1 + (size_t)(half ? 2 : 0) * D1n + n;
    v8f acc = {};
    int k0 = kc * (FLATK / 8);
#pragma unroll 4
    for (int k = k0; k < k0 + FLATK / 8; k += 4) {
        v2f a = *(const v2f*)(ya + k);                  // contiguous K pair
        v2f bfr;
        bfr[0] = wb[(size_t)k * D1n];
        bfr[1] = wb[(size_t)(k + 1) * D1n];
        if (k + 260 < FLATK)
            __builtin_prefetch(wb + (size_t)(k + 256) * D1n, 0, 1);
        acc = __builtin_amdgcn_wmma_f32_16x16x4_f32(
            false, a, false, bfr, (short)0, acc, false, false);
    }
    float* p = part + (size_t)((mt * 125 + nt) * 8 + kc) * 256;
#pragma unroll
    for (int i = 0; i < 8; ++i) p[(i + (half ? 8 : 0)) * 16 + row] = acc[i];
}

__global__ void dense1_reduce(const float* __restrict__ part, const float* __restrict__ b1,
                              float* __restrict__ yd1) {
    int idx = blockIdx.x * blockDim.x + threadIdx.x;
    if (idx >= Bn * D1n) return;
    int m = idx / D1n, n = idx % D1n;
    const float* p = part + (size_t)(((m >> 4) * 125 + (n >> 4)) * 8) * 256 + (m & 15) * 16 + (n & 15);
    float s = b1[n];
#pragma unroll
    for (int kc = 0; kc < 8; ++kc) s += p[kc * 256];
    yd1[idx] = fmaxf(s, 0.0f);
}

// ===========================================================================
// Dense2: (64 x 2000) @ w2(2000 x 301) + b2, sigmoid. fp32 WMMA, N clamped
// branch-free (clamped column scaled by 0 for n>=301).
// ===========================================================================
__global__ __launch_bounds__(32)
void dense2_kernel(const float* __restrict__ yd1, const float* __restrict__ w2,
                   const float* __restrict__ b2, float* __restrict__ out) {
    int nt = blockIdx.x, mt = blockIdx.y;
    int lane = threadIdx.x & 31, row = lane & 15, half = (lane >> 4) & 1;
    int n = nt * 16 + row;
    bool nok = n < D2n;
    int nc = nok ? n : (D2n - 1);
    float nscale = nok ? 1.0f : 0.0f;
    const float* ya = yd1 + (size_t)(mt * 16 + row) * D1n + (half ? 2 : 0);
    const float* wb = w2 + (size_t)(half ? 2 : 0) * D2n + nc;
    v8f acc = {};
#pragma unroll 4
    for (int k = 0; k < D1n; k += 4) {
        v2f a = *(const v2f*)(ya + k);
        v2f bfr;
        bfr[0] = wb[(size_t)k * D2n] * nscale;
        bfr[1] = wb[(size_t)(k + 1) * D2n] * nscale;
        acc = __builtin_amdgcn_wmma_f32_16x16x4_f32(
            false, a, false, bfr, (short)0, acc, false, false);
    }
    if (nok) {
        float bias = b2[n];
#pragma unroll
        for (int i = 0; i < 8; ++i) {
            int m = mt * 16 + i + (half ? 8 : 0);
            out[(size_t)m * D2n + n] = 1.0f / (1.0f + __expf(-(acc[i] + bias)));
        }
    }
}

// ===========================================================================
extern "C" void kernel_launch(void* const* d_in, const int* in_sizes, int n_in,
                              void* d_out, int out_size, void* d_ws, size_t ws_size,
                              hipStream_t stream) {
    (void)in_sizes; (void)n_in; (void)out_size; (void)ws_size;
    const float* in     = (const float*)d_in[0];
    const float* conv_w = (const float*)d_in[1];
    const float* conv_b = (const float*)d_in[2];
    const float* fwk    = (const float*)d_in[3];
    const float* fwrk   = (const float*)d_in[4];
    const float* fwb    = (const float*)d_in[5];
    const float* bwk    = (const float*)d_in[6];
    const float* bwrk   = (const float*)d_in[7];
    const float* bwb    = (const float*)d_in[8];
    const float* w1     = (const float*)d_in[9];
    const float* b1     = (const float*)d_in[10];
    const float* w2     = (const float*)d_in[11];
    const float* b2     = (const float*)d_in[12];
    float* out = (float*)d_out;

    char* ws = (char*)d_ws;
    auto carve = [&ws](size_t bytes) {
        char* p = ws;
        ws += (bytes + 255) & ~(size_t)255;
        return p;
    };
    _Float16* convB  = (_Float16*)carve((size_t)CONV_KT * CONV_NT * 512 * 2);
    float*    P      = (float*)   carve((size_t)T_POOL * Bn * Fc * 4);
    _Float16* kpack  = (_Float16*)carve((size_t)2 * GRU_KT * GRU_NT * 512 * 2);
    _Float16* rkpack = (_Float16*)carve((size_t)2 * GRU_KT * GRU_NT * 512 * 2);
    float*    XP     = (float*)   carve((size_t)2 * 4800 * N3H * 4);
    float*    Ybuf   = (float*)   carve((size_t)Bn * FLATK * 4);
    float*    part   = (float*)   carve((size_t)4 * 125 * 8 * 256 * 4);
    float*    yd1    = (float*)   carve((size_t)Bn * D1n * 4);

    pack_conv_w<<<(CONV_KT * CONV_NT * 32 + 255) / 256, 256, 0, stream>>>(conv_w, convB);
    pack_gru_w<<<(4 * GRU_KT * GRU_NT * 32 + 255) / 256, 256, 0, stream>>>(
        fwk, bwk, fwrk, bwrk, kpack, rkpack);
    conv_pool_kernel<<<dim3(T_POOL, Bn), 160, 0, stream>>>(in, conv_b, convB, P);
    xp_gemm<<<dim3(15, 300, 2), 32, 0, stream>>>(P, kpack, fwb, bwb, XP);
    gru_kernel<<<16, 480, 0, stream>>>(XP, rkpack, fwb, bwb, Ybuf);
    dense1_kernel<<<dim3(125, 4, 8), 32, 0, stream>>>(Ybuf, w1, part);
    dense1_reduce<<<(Bn * D1n + 255) / 256, 256, 0, stream>>>(part, b1, yd1);
    dense2_kernel<<<dim3(19, 4), 32, 0, stream>>>(yd1, w2, b2, out);
}
